// minerva_detEx_56135222559425
// MI455X (gfx1250) — matrix operand order repaired and checked by reading, not compiled, not run on hardware
//
#include <hip/hip_runtime.h>

typedef unsigned short u16;
typedef unsigned int   u32;
typedef unsigned long long u64;

typedef __attribute__((ext_vector_type(16))) __bf16 v16bf;
typedef __attribute__((ext_vector_type(8)))  float  v8f;

union Frag32 { v16bf v; uint4 q[2]; };

__device__ __forceinline__ u16 f2bf(float f) {
  // round-to-nearest-even fp32 -> bf16
  u32 u = __float_as_uint(f);
  u += 0x7fffu + ((u >> 16) & 1u);
  return (u16)(u >> 16);
}

// ---------------------------------------------------------------------------
// fp32 -> bf16 conversion, 2 elements per thread packed into one dword store
// ---------------------------------------------------------------------------
__global__ void cvt_pair_k(const float* __restrict__ src, u32* __restrict__ dst, int npair) {
  int i = blockIdx.x * blockDim.x + threadIdx.x;
  if (i < npair) {
    float2 f = ((const float2*)src)[i];
    dst[i] = (u32)f2bf(f.x) | ((u32)f2bf(f.y) << 16);
  }
}

// ---------------------------------------------------------------------------
// exCT[c][d] = bf16( sum_j ex_classes[d][j] * class_reps[j][c] ), padded to 32 rows
// ---------------------------------------------------------------------------
__global__ void build_exCT_k(const float* __restrict__ exCls, const float* __restrict__ reps,
                             u16* __restrict__ exCT, int D, int C) {
  int i = blockIdx.x * blockDim.x + threadIdx.x;
  if (i >= 32 * D) return;
  int c = i / D, d = i - c * D;
  float v = 0.f;
  if (c < C) {
    for (int j = 0; j < C; ++j) v += exCls[(size_t)d * C + j] * reps[j * C + c];
  }
  exCT[(size_t)c * D + d] = f2bf(v);
}

// ---------------------------------------------------------------------------
// Fused projection: out = l2norm_rows( Xb @ gWb^T + gb ) in bf16.
// Block: 16 rows x H cols (H = 1024). 8 waves split N (128 cols each, 8 tiles).
// A tile staged in LDS (shared); B fragments double-buffered in registers so
// the global load for tile t+1 overlaps the WMMA of tile t.
// ---------------------------------------------------------------------------
__global__ __launch_bounds__(256) void proj_norm_wmma(
    const u16* __restrict__ A,   // [M,K] bf16
    const u16* __restrict__ Bw,  // [H,K] bf16 (gW row-major: B[k][n] = gW[n][k])
    const float* __restrict__ gb,
    u16* __restrict__ out,       // [M,H] bf16 normalized
    int K, int H)
{
  constexpr int TN  = 8;
  constexpr int LST = 40;                 // 32 + 8 pad (80B rows, 16B-aligned)
  __shared__ __align__(16) u16 ldsA[16 * LST];
  __shared__ float ldsSS[16];

  const int tid  = threadIdx.x;
  const int wave = tid >> 5;
  const int lane = tid & 31;
  const int l16  = lane & 15;
  const int h    = lane >> 4;
  const int mBase = blockIdx.x * 16;
  const int nBase = wave * 128;

  if (tid < 16) ldsSS[tid] = 0.f;

  v8f acc[TN] = {};

  for (int k0 = 0; k0 < K; k0 += 32) {
    if (tid < 64) {
      int row = tid >> 2, c = tid & 3;
      uint4 dq = *(const uint4*)(A + (size_t)(mBase + row) * K + k0 + c * 8);
      *(uint4*)(&ldsA[row * LST + c * 8]) = dq;
    }
    __syncthreads();
    Frag32 a;                              // A 16x32: elems 0..7 K=8h.., 8..15 K=16+8h..
    const u16* ap = &ldsA[l16 * LST + 8 * h];
    a.q[0] = *(const uint4*)(ap);
    a.q[1] = *(const uint4*)(ap + 16);

    const u16* bbase = Bw + (size_t)(nBase + l16) * K + k0 + 8 * h;
    Frag32 bf[2];
    bf[0].q[0] = *(const uint4*)(bbase);
    bf[0].q[1] = *(const uint4*)(bbase + 16);
    for (int t = 0; t < TN; ++t) {
      if (t + 1 < TN) {                    // prefetch next B fragment
        const u16* bp = bbase + (size_t)(t + 1) * 16 * K;
        bf[(t + 1) & 1].q[0] = *(const uint4*)(bp);
        bf[(t + 1) & 1].q[1] = *(const uint4*)(bp + 16);
      }
      acc[t] = __builtin_amdgcn_wmma_f32_16x16x32_bf16(
          false, a.v, false, bf[t & 1].v, (short)0, acc[t], false, false);
    }
    __syncthreads();
  }

  float gbv[TN];
  for (int t = 0; t < TN; ++t) gbv[t] = gb[nBase + t * 16 + l16];

  for (int r = 0; r < 8; ++r) {            // C/D: M = r + 8h, N = l16
    float ss = 0.f;
    for (int t = 0; t < TN; ++t) { float v = acc[t][r] + gbv[t]; ss += v * v; }
    atomicAdd(&ldsSS[r + 8 * h], ss);      // ds_add_f32 across 8 waves
  }
  __syncthreads();
  for (int r = 0; r < 8; ++r) {
    const int m = mBase + r + 8 * h;
    const float scale = 1.0f / fmaxf(sqrtf(ldsSS[r + 8 * h]), 1e-12f);
    for (int t = 0; t < TN; ++t) {
      int n = nBase + t * 16 + l16;
      out[(size_t)m * H + n] = f2bf((acc[t][r] + gbv[t]) * scale);
    }
  }
}

// ---------------------------------------------------------------------------
// Big NT-GEMM with async global->LDS double buffering (CDNA5 ASYNC path):
// C = A[M,K] @ B[N,K]^T, bf16 in, f32 accumulate, block = 128x128, 8 waves
// split M. B tiles (128 rows x 64 K) are copied with
// GLOBAL_LOAD_ASYNC_TO_LDS_B128 into LDS buffer (p^1) while WMMAs consume
// buffer p; s_wait_asynccnt 0x4 retires only the current stage (in-order).
// EPI 0: store f32.  EPI 1: store bf16(x^3)  (power-sign activation, p=3).
// ---------------------------------------------------------------------------
template<int EPI>
__global__ __launch_bounds__(256) void gemm_nt_async_wmma(
    const u16* __restrict__ A, const u16* __restrict__ Bm, void* __restrict__ Cv,
    int K, int lda, int ldb, int ldc)
{
  constexpr int LST = 72;                  // 64 + 8 pad (144B rows, 16B-aligned)
  __shared__ __align__(16) u16 ldsB[2][128 * LST];

  const int tid  = threadIdx.x;
  const int wave = tid >> 5;
  const int lane = tid & 31;
  const int l16  = lane & 15;
  const int h    = lane >> 4;
  const int mBase = blockIdx.y * 128 + wave * 16;
  const int nBase = blockIdx.x * 128;

  v8f acc[8] = {};
  const u16* aRow = A + (size_t)(mBase + l16) * lda;

  // per-thread async copy of one 128x64 bf16 B-stage (4 x b128 per thread)
  auto issueStage = [&](int buf, int k0) {
#pragma unroll
    for (int j = 0; j < 4; ++j) {
      int c = tid + 256 * j;               // 1024 16B-chunks total
      int row = c >> 3, q = c & 7;
      const u16* gp = Bm + (size_t)(nBase + row) * ldb + k0 + q * 8;
      u32 lo = (u32)(uintptr_t)&ldsB[buf][row * LST + q * 8];
      u64 ga = (u64)(uintptr_t)gp;
      asm volatile("global_load_async_to_lds_b128 %0, %1, off"
                   :: "v"(lo), "v"(ga) : "memory");
    }
  };

  issueStage(0, 0);

  for (int k0 = 0; k0 < K; k0 += 64) {
    const int cur = (k0 >> 6) & 1;
    __syncthreads();                       // everyone done reading buf cur^1
    if (k0 + 64 < K) {
      issueStage(cur ^ 1, k0 + 64);        // overlap next copy with this compute
      asm volatile("s_wait_asynccnt 0x4" ::: "memory");  // retire current stage only
    } else {
      asm volatile("s_wait_asynccnt 0x0" ::: "memory");
    }
    __syncthreads();                       // buf cur visible to all waves

#pragma unroll
    for (int s = 0; s < 2; ++s) {          // two K=32 sub-steps
      Frag32 a;
      const u16* ap = aRow + k0 + 32 * s;
      a.q[0] = *(const uint4*)(ap + 8 * h);
      a.q[1] = *(const uint4*)(ap + 16 + 8 * h);
#pragma unroll
      for (int t = 0; t < 8; ++t) {
        Frag32 b;
        const u16* bp = &ldsB[cur][(t * 16 + l16) * LST + 32 * s + 8 * h];
        b.q[0] = *(const uint4*)(bp);
        b.q[1] = *(const uint4*)(bp + 16);
        acc[t] = __builtin_amdgcn_wmma_f32_16x16x32_bf16(
            false, a.v, false, b.v, (short)0, acc[t], false, false);
      }
    }
  }

#pragma unroll
  for (int t = 0; t < 8; ++t) {
    const int n = nBase + t * 16 + l16;
    for (int r = 0; r < 8; ++r) {
      const int m = mBase + r + 8 * h;
      float x = acc[t][r];
      if constexpr (EPI == 1) {
        ((u16*)Cv)[(size_t)m * ldc + n] = f2bf(x * x * x);  // sign(s)|s|^3 == s^3
      } else {
        ((float*)Cv)[(size_t)m * ldc + n] = x;
      }
    }
  }
}

// ---------------------------------------------------------------------------
// Thin NT-GEMM (N = 32, memory-bound): synchronous LDS staging is fine here.
// ---------------------------------------------------------------------------
__global__ __launch_bounds__(256) void gemm_nt_thin_wmma(
    const u16* __restrict__ A, const u16* __restrict__ Bm, float* __restrict__ Cf,
    int K, int lda, int ldb, int ldc)
{
  constexpr int BN  = 32;
  constexpr int TN  = 2;
  constexpr int LST = 40;
  __shared__ __align__(16) u16 ldsB[BN * LST];

  const int tid  = threadIdx.x;
  const int wave = tid >> 5;
  const int lane = tid & 31;
  const int l16  = lane & 15;
  const int h    = lane >> 4;
  const int mBase = blockIdx.y * 128 + wave * 16;
  const int nBase = blockIdx.x * BN;

  v8f acc[TN] = {};
  const u16* aRow = A + (size_t)(mBase + l16) * lda;

  for (int k0 = 0; k0 < K; k0 += 32) {
    if (tid < BN * 4) {
      int row = tid >> 2, c = tid & 3;
      uint4 dq = *(const uint4*)(Bm + (size_t)(nBase + row) * ldb + k0 + c * 8);
      *(uint4*)(&ldsB[row * LST + c * 8]) = dq;
    }
    __syncthreads();
    Frag32 a;
    a.q[0] = *(const uint4*)(aRow + k0 + 8 * h);
    a.q[1] = *(const uint4*)(aRow + k0 + 16 + 8 * h);
    for (int t = 0; t < TN; ++t) {
      Frag32 b;
      const u16* bp = &ldsB[(t * 16 + l16) * LST + 8 * h];
      b.q[0] = *(const uint4*)(bp);
      b.q[1] = *(const uint4*)(bp + 16);
      acc[t] = __builtin_amdgcn_wmma_f32_16x16x32_bf16(
          false, a.v, false, b.v, (short)0, acc[t], false, false);
    }
    __syncthreads();
  }

  for (int t = 0; t < TN; ++t) {
    const int n = nBase + t * 16 + l16;
    for (int r = 0; r < 8; ++r) {
      const int m = mBase + r + 8 * h;
      Cf[(size_t)m * ldc + n] = acc[t][r];
    }
  }
}

// ---------------------------------------------------------------------------
// logits[b][j] = -sqrt( sum_c (echo[b][c] - class_reps[j][c])^2 )
// ---------------------------------------------------------------------------
__global__ void logits_k(const float* __restrict__ echo, const float* __restrict__ reps,
                         float* __restrict__ out, int C) {
  int b = blockIdx.x, j = threadIdx.x;
  if (j >= C) return;
  float d2 = 0.f;
  for (int c = 0; c < C; ++c) {
    float diff = echo[(size_t)b * 32 + c] - reps[j * C + c];
    d2 += diff * diff;
  }
  out[(size_t)b * C + j] = -sqrtf(d2);
}

// ---------------------------------------------------------------------------
extern "C" void kernel_launch(void* const* d_in, const int* in_sizes, int n_in,
                              void* d_out, int out_size, void* d_ws, size_t ws_size,
                              hipStream_t stream) {
  const float* features = (const float*)d_in[0];  // [B,H]
  const float* exfeat   = (const float*)d_in[1];  // [D,H]
  const float* gW       = (const float*)d_in[2];  // [H,H]
  const float* gb       = (const float*)d_in[3];  // [H]
  const float* exCls    = (const float*)d_in[4];  // [D,C]
  const float* reps     = (const float*)d_in[5];  // [C,C]

  const int H = in_sizes[3];        // 1024
  const int B = in_sizes[0] / H;    // 4096
  const int D = in_sizes[1] / H;    // 16384
  const int C = in_sizes[4] / D;    // 28

  char* ws = (char*)d_ws;
  size_t off = 0;
  auto carve = [&](size_t bytes) -> char* {
    char* p = ws + off;
    off = (off + bytes + 255) & ~(size_t)255;
    return p;
  };
  u16*   featB = (u16*)carve((size_t)B * H * 2);
  u16*   exB   = (u16*)carve((size_t)D * H * 2);
  u16*   gWb   = (u16*)carve((size_t)H * H * 2);
  u16*   fn    = (u16*)carve((size_t)B * H * 2);   // normalized f, bf16
  u16*   en    = (u16*)carve((size_t)D * H * 2);   // normalized ex, bf16
  u16*   Acu   = (u16*)carve((size_t)B * D * 2);   // cubed similarities, bf16
  u16*   exCT  = (u16*)carve((size_t)32 * D * 2);  // class indicators^T, padded
  float* echo  = (float*)carve((size_t)B * 32 * 4);
  (void)ws_size; (void)n_in;

  // 1) fp32 -> bf16 conversions
  { int np = (B * H) / 2; cvt_pair_k<<<dim3((np + 255) / 256), dim3(256), 0, stream>>>(features, (u32*)featB, np); }
  { int np = (D * H) / 2; cvt_pair_k<<<dim3((np + 255) / 256), dim3(256), 0, stream>>>(exfeat,   (u32*)exB,   np); }
  { int np = (H * H) / 2; cvt_pair_k<<<dim3((np + 255) / 256), dim3(256), 0, stream>>>(gW,       (u32*)gWb,   np); }

  // 2) fused Linear + bias + row L2-normalize (WMMA)
  proj_norm_wmma<<<dim3(B / 16), dim3(256), 0, stream>>>(featB, gWb, gb, fn, H, H);
  proj_norm_wmma<<<dim3(D / 16), dim3(256), 0, stream>>>(exB,   gWb, gb, en, H, H);

  // 3) exemplar class reps, transposed bf16 [32, D]
  build_exCT_k<<<dim3((32 * D + 255) / 256), dim3(256), 0, stream>>>(exCls, reps, exCT, D, C);

  // 4) big cosine-similarity GEMM (async double-buffered), cube epilogue -> bf16 [B, D]
  gemm_nt_async_wmma<1><<<dim3(D / 128, B / 128), dim3(256), 0, stream>>>(
      fn, en, (void*)Acu, H, H, H, D);

  // 5) echo = a @ exC  (thin GEMM, N padded to 32) -> f32 [B, 32]
  gemm_nt_thin_wmma<<<dim3(1, B / 128), dim3(256), 0, stream>>>(
      Acu, exCT, echo, D, D, D, 32);

  // 6) logits = -cdist(echo, class_reps)
  logits_k<<<dim3(B), dim3(32), 0, stream>>>(echo, reps, (float*)d_out, C);
  (void)out_size;
}
